// BiDirectionalRNN_32779190403214
// MI455X (gfx1250) — compile-verified, matrix operand order checked by reading, not run on hardware
//
#include <hip/hip_runtime.h>

// ---------------- problem constants ----------------
static constexpr int B   = 128;
static constexpr int T   = 256;
static constexpr int IN  = 256;
static constexpr int H   = 512;
static constexpr int G4  = 4 * H;     // 2048
static constexpr int OUT = 256;

// LDS staging geometry: per K-chunk (32 K-elements):
//   A tile: 64 rows x 32  -> rows 0..63
//   B tile: 256 rows x 32 -> rows 64..319  (4 gates x 64 cols)
// Row pitch padded to 40 elements (80B) => the 16 lanes of a fragment read
// hit 16 distinct 16B-aligned slots mod 256B (conflict-free).
static constexpr int KC      = 32;
static constexpr int PITCH   = 40;
static constexpr int ROWS    = 64 + 256;             // 320 rows / chunk
static constexpr int BUFELEM = ROWS * PITCH;         // 12800 bf16 / buffer
static constexpr int SEG_PER_THREAD = 5;             // 1280 16B segs / 256 thr

// ---------------- types ----------------
typedef __bf16 bf16_t;
typedef __attribute__((ext_vector_type(16))) __bf16 bf16x16;
typedef __attribute__((ext_vector_type(8)))  __bf16 bf16x8;
typedef __attribute__((ext_vector_type(8)))  float  f32x8;

#define WMMA_BF16(a, b, c) \
  __builtin_amdgcn_wmma_f32_16x16x32_bf16(false, (a), false, (b), (short)0, (c), false, false)

// ---------------- CDNA5 async global->LDS path ----------------
// Emitted via inline asm (ISA §15.18.3 opcode 98); the toolchain builtin has a
// descriptor-style signature we don't rely on. Generic shared pointers are
// {aperture_hi, lds_offset}; low 32 bits are the LDS byte address.
#if defined(__has_builtin)
#if __has_builtin(__builtin_amdgcn_s_wait_asynccnt)
#define WAIT_ASYNC(n) __builtin_amdgcn_s_wait_asynccnt(n)
#else
#define WAIT_ASYNC(n) asm volatile("s_wait_asynccnt %0" ::"i"(n) : "memory")
#endif
#else
#define WAIT_ASYNC(n) asm volatile("s_wait_asynccnt %0" ::"i"(n) : "memory")
#endif

// Copy 16 bytes global -> LDS (async DMA; per-lane addresses, 1 instr/wave).
__device__ __forceinline__ void copy16(bf16_t* ldst, const bf16_t* gsrc) {
  unsigned int       loff = (unsigned int)(unsigned long long)(void*)ldst;
  unsigned long long ga   = (unsigned long long)(const void*)gsrc;
  asm volatile("global_load_async_to_lds_b128 %0, %1, off"
               ::"v"(loff), "v"(ga)
               : "memory");
}

// ---------------- fragment loader ----------------
// 16x32 bf16 fragment (A layout; B operand = same layout applied to row-major
// W, i.e. B^T). base points at (tile_row0, k0); ld = leading dim in elements
// (32-bit: all offsets fit, avoids 64-bit VALU address math).
__device__ __forceinline__ bf16x16 load_frag(const bf16_t* base, unsigned ld) {
  const unsigned lane = threadIdx.x & 31;
  const bf16_t* p = base + (lane & 15) * ld + ((lane >> 4) << 3);
  bf16x8 lo = *(const bf16x8*)(p);
  bf16x8 hi = *(const bf16x8*)(p + 16);
  bf16x16 f;
#pragma unroll
  for (int e = 0; e < 8; ++e) { f[e] = lo[e]; f[8 + e] = hi[e]; }
  return f;
}

__device__ __forceinline__ float sigf(float x) { return 1.0f / (1.0f + __expf(-x)); }

// Stage one K-chunk (A tile + 4-gate B panel) into an LDS buffer.
// Segment->thread mapping is uniform per phase: threads' first segment covers
// the whole A tile (rows 0..63), remaining 4 segments cover the B panel
// (rows 64..319) -- no divergence, pure 32-bit address math.
__device__ __forceinline__ void stage_chunk(bf16_t* buf,
                                            const bf16_t* __restrict__ Abase, unsigned Ald,
                                            int tile_m,
                                            const bf16_t* __restrict__ W, unsigned K,
                                            int tile_j, int kk, int tid) {
  {
    const int row = tid >> 2;
    const int seg = (tid & 3) * 8;
    copy16(buf + row * PITCH + seg,
           Abase + (unsigned)(tile_m + row) * Ald + (unsigned)(kk + seg));
  }
#pragma unroll
  for (int it = 1; it < SEG_PER_THREAD; ++it) {
    const int q   = tid + it * 256;          // 256..1279
    const int row = q >> 2;                  // 64..319
    const int seg = (q & 3) * 8;
    const int br  = row - 64;
    const unsigned wrow = (unsigned)((br >> 6) * H + tile_j + (br & 63));
    copy16(buf + row * PITCH + seg, W + wrow * K + (unsigned)(kk + seg));
  }
}

// ---------------- step kernel: 4 LSTM cells per launch ----------------
// blockIdx.x: [cell(2b) | sub(4b)] -> 64 blocks, 256 threads (8 waves).
// cell: 0=L0 fwd, 1=L0 rev, 2=L1 fwd, 3=L1 rev. Layer1 pipelined 1 step back.
__global__ __launch_bounds__(256) void lstm_step(
    const bf16_t* __restrict__ xbf,
    const bf16_t* __restrict__ Wih0, const bf16_t* __restrict__ Whh0,
    const bf16_t* __restrict__ Wih1, const bf16_t* __restrict__ Whh1,
    const float* __restrict__ bih0, const float* __restrict__ bhh0,
    const float* __restrict__ bih1, const float* __restrict__ bhh1,
    bf16_t* __restrict__ h0, bf16_t* __restrict__ h1,   // [dir][parity][B*H]
    float* __restrict__ c0, float* __restrict__ c1,     // [dir][B*H]
    int t0, int e0, int t1, int e1) {
  __shared__ alignas(16) bf16_t smem[2 * BUFELEM];      // 50 KB, double buffer

  const int cell  = blockIdx.x >> 4;
  const int layer = cell >> 1;
  const int dir   = cell & 1;
  if (layer == 0 && !e0) return;
  if (layer == 1 && !e1) return;
  const int t = layer ? t1 : t0;

  const int sub    = blockIdx.x & 15;
  const int tile_m = (sub >> 3) * 64;   // batch-row tile (2)
  const int tile_j = (sub & 7) * 64;    // hidden-col tile (8)

  // --- operand selection ---
  const bf16_t* A1; unsigned A1ld; int K1; const bf16_t* W1;
  if (layer == 0) {
    const int tt = dir ? ((T - t) & (T - 1)) : t;   // reverse stream index
    A1 = xbf + (size_t)tt * IN;  A1ld = (unsigned)(T * IN);  K1 = IN;  W1 = Wih0;
  } else {
    A1 = h0 + (size_t)(dir * 2 + ((t + 1) & 1)) * B * H;   // h0^(t)
    A1ld = H;  K1 = H;  W1 = Wih1;
  }
  bf16_t* hs = layer ? h1 : h0;
  const bf16_t* A2   = hs + (size_t)(dir * 2 + (t & 1)) * B * H;        // h prev
  bf16_t*       hout = hs + (size_t)(dir * 2 + ((t + 1) & 1)) * B * H;  // h new
  const bf16_t* W2   = layer ? Whh1 : Whh0;
  float*        c    = (layer ? c1 : c0) + (size_t)dir * B * H;
  const float*  bih  = layer ? bih1 : bih0;
  const float*  bhh  = layer ? bhh1 : bhh0;

  const int tid = threadIdx.x;
  const int w = tid >> 5, lane = tid & 31;
  const int sm = w >> 1, sn_base = (w & 1) * 2;

  const int nch1 = K1 / KC;
  const int nch  = nch1 + H / KC;
  auto chunk_src = [&](int ci, const bf16_t*& Ab, unsigned& ald,
                       const bf16_t*& Wb, int& Kb, int& kk) {
    if (ci < nch1) { Ab = A1; ald = A1ld; Wb = W1; Kb = K1; kk = ci * KC; }
    else           { Ab = A2; ald = H;    Wb = W2; Kb = H;  kk = (ci - nch1) * KC; }
  };

  const f32x8 vzero = {0.f, 0.f, 0.f, 0.f, 0.f, 0.f, 0.f, 0.f};
  f32x8 acc[4][2];
#pragma unroll
  for (int g = 0; g < 4; ++g)
#pragma unroll
    for (int s = 0; s < 2; ++s) acc[g][s] = vzero;

  // prologue: stage chunk 0
  {
    const bf16_t* Ab; unsigned ald; const bf16_t* Wb; int Kb, kk;
    chunk_src(0, Ab, ald, Wb, Kb, kk);
    stage_chunk(smem, Ab, ald, tile_m, Wb, (unsigned)Kb, tile_j, kk, tid);
  }

  for (int ci = 0; ci < nch; ++ci) {
    __syncthreads();                       // all waves done with buf[(ci+1)&1]
    if (ci + 1 < nch) {                    // prefetch next chunk (stays in flight
      const bf16_t* Ab; unsigned ald; const bf16_t* Wb; int Kb, kk;
      chunk_src(ci + 1, Ab, ald, Wb, Kb, kk);
      stage_chunk(smem + ((ci + 1) & 1) * BUFELEM, Ab, ald, tile_m,
                  Wb, (unsigned)Kb, tile_j, kk, tid);
      WAIT_ASYNC(SEG_PER_THREAD);          // in-order => chunk ci has landed
    } else {
      WAIT_ASYNC(0);
    }
    __syncthreads();                       // chunk ci visible to all waves

    const bf16_t* buf = smem + (ci & 1) * BUFELEM;
    bf16x16 a = load_frag(buf + (sm * 16) * PITCH, PITCH);

    // software-pipelined B fragments: load frag idx+1 before WMMA idx so the
    // LDS latency hides under the previous matrix op (distinct registers,
    // no per-WMMA s_wait_dscnt 0 stall).
    auto load_b = [&](int idx) {
      const int g = idx >> 1, s = idx & 1;
      const int brow = 64 + g * 64 + (sn_base + s) * 16;
      return load_frag(buf + brow * PITCH, PITCH);
    };
    bf16x16 bcur = load_b(0);
#pragma unroll
    for (int idx = 0; idx < 8; ++idx) {
      bf16x16 bnxt = bcur;
      if (idx + 1 < 8) bnxt = load_b(idx + 1);
      acc[idx >> 1][idx & 1] = WMMA_BF16(a, bcur, acc[idx >> 1][idx & 1]);
      bcur = bnxt;
    }
  }

  // --- fused LSTM epilogue: all four gates live in this lane ---
  const int rbase = tile_m + sm * 16 + ((lane >> 4) << 3);
#pragma unroll
  for (int s = 0; s < 2; ++s) {
    const int j = tile_j + (sn_base + s) * 16 + (lane & 15);
    float bi[4];
#pragma unroll
    for (int g = 0; g < 4; ++g) { int n = g * H + j; bi[g] = bih[n] + bhh[n]; }
#pragma unroll
    for (int r = 0; r < 8; ++r) {
      const size_t idx = (size_t)(rbase + r) * H + j;
      float iv = sigf(acc[0][s][r] + bi[0]);
      float fv = sigf(acc[1][s][r] + bi[1]);
      float gv = tanhf(acc[2][s][r] + bi[2]);
      float ov = sigf(acc[3][s][r] + bi[3]);
      float cn = fv * c[idx] + iv * gv;
      c[idx]    = cn;
      hout[idx] = (bf16_t)(ov * tanhf(cn));
    }
  }
}

// ---------------- final FC: out[128x256] = [h_f | h_r] @ Wfc^T + b ----------------
__global__ __launch_bounds__(256) void fc_kernel(
    const bf16_t* __restrict__ hf, const bf16_t* __restrict__ hr,
    const bf16_t* __restrict__ Wfc, const float* __restrict__ bfc,
    float* __restrict__ out) {
  const int tile_m = (blockIdx.x >> 2) * 64;
  const int tile_n = (blockIdx.x & 3) * 64;
  const int w = threadIdx.x >> 5, lane = threadIdx.x & 31;
  const int sm = w >> 1, sn_base = (w & 1) * 2;
  const int row0 = tile_m + sm * 16;

  const f32x8 vzero = {0.f, 0.f, 0.f, 0.f, 0.f, 0.f, 0.f, 0.f};
  f32x8 acc[2] = {vzero, vzero};

  for (int kk = 0; kk < 2 * H; kk += 32) {
    const bf16_t* ab = (kk < H) ? (hf + kk) : (hr + (kk - H));
    bf16x16 a = load_frag(ab + (unsigned)row0 * H, H);
#pragma unroll
    for (int s = 0; s < 2; ++s) {
      int n = tile_n + (sn_base + s) * 16;
      bf16x16 b = load_frag(Wfc + (unsigned)n * (2 * H) + kk, 2 * H);
      acc[s] = WMMA_BF16(a, b, acc[s]);
    }
  }
  const int rbase = row0 + ((lane >> 4) << 3);
#pragma unroll
  for (int s = 0; s < 2; ++s) {
    const int col = tile_n + (sn_base + s) * 16 + (lane & 15);
#pragma unroll
    for (int r = 0; r < 8; ++r)
      out[(size_t)(rbase + r) * OUT + col] = acc[s][r] + bfc[col];
  }
}

// ---------------- utility kernels ----------------
__global__ void f32_to_bf16(const float* __restrict__ s, bf16_t* __restrict__ d, int n) {
  int i = blockIdx.x * blockDim.x + threadIdx.x;
  int st = gridDim.x * blockDim.x;
  for (; i < n; i += st) d[i] = (bf16_t)s[i];
}
__global__ void zero_u32(unsigned int* __restrict__ p, int n) {
  int i = blockIdx.x * blockDim.x + threadIdx.x;
  int st = gridDim.x * blockDim.x;
  for (; i < n; i += st) p[i] = 0u;
}

// ---------------- host launch ----------------
extern "C" void kernel_launch(void* const* d_in, const int* in_sizes, int n_in,
                              void* d_out, int out_size, void* d_ws, size_t ws_size,
                              hipStream_t stream) {
  const float* x     = (const float*)d_in[0];
  const float* W_ih0 = (const float*)d_in[1];
  const float* W_hh0 = (const float*)d_in[2];
  const float* b_ih0 = (const float*)d_in[3];
  const float* b_hh0 = (const float*)d_in[4];
  const float* W_ih1 = (const float*)d_in[5];
  const float* W_hh1 = (const float*)d_in[6];
  const float* b_ih1 = (const float*)d_in[7];
  const float* b_hh1 = (const float*)d_in[8];
  const float* W_fc  = (const float*)d_in[9];
  const float* b_fc  = (const float*)d_in[10];
  float* out = (float*)d_out;

  char* ws = (char*)d_ws;
  size_t off = 0;
  auto carve = [&](size_t bytes) -> char* {
    char* p = ws + off;
    off += (bytes + 255) & ~(size_t)255;
    return p;
  };
  bf16_t* xbf   = (bf16_t*)carve((size_t)B * T * IN * 2);
  bf16_t* Wih0b = (bf16_t*)carve((size_t)G4 * IN * 2);
  bf16_t* Whh0b = (bf16_t*)carve((size_t)G4 * H * 2);
  bf16_t* Wih1b = (bf16_t*)carve((size_t)G4 * H * 2);
  bf16_t* Whh1b = (bf16_t*)carve((size_t)G4 * H * 2);
  bf16_t* Wfcb  = (bf16_t*)carve((size_t)OUT * 2 * H * 2);
  char*   state = ws + off;
  bf16_t* h0    = (bf16_t*)carve((size_t)4 * B * H * 2);  // [dir][parity]
  bf16_t* h1    = (bf16_t*)carve((size_t)4 * B * H * 2);
  float*  c0    = (float*)carve((size_t)2 * B * H * 4);   // [dir]
  float*  c1    = (float*)carve((size_t)2 * B * H * 4);
  const size_t stateWords = ((char*)(c1 + 2 * B * H) - state) / 4;

  // 1) zero h/c state (ws is poisoned; must be deterministic per call)
  zero_u32<<<512, 256, 0, stream>>>((unsigned int*)state, (int)stateWords);

  // 2) bf16 conversions (weights are L2-resident afterwards)
  auto cvt = [&](const float* s, bf16_t* d, int n) {
    int blocks = (n + 255) / 256; if (blocks > 1024) blocks = 1024;
    f32_to_bf16<<<blocks, 256, 0, stream>>>(s, d, n);
  };
  cvt(x,     xbf,   B * T * IN);
  cvt(W_ih0, Wih0b, G4 * IN);
  cvt(W_hh0, Whh0b, G4 * H);
  cvt(W_ih1, Wih1b, G4 * H);
  cvt(W_hh1, Whh1b, G4 * H);
  cvt(W_fc,  Wfcb,  OUT * 2 * H);

  // 3) recurrence: layer1 pipelined one step behind layer0; both directions
  //    run concurrently inside each launch (4 cells x 16 tiles = 64 blocks).
  for (int k = 0; k <= T; ++k) {
    const int e0 = (k < T) ? 1 : 0;
    const int e1 = (k > 0) ? 1 : 0;
    lstm_step<<<64, 256, 0, stream>>>(xbf, Wih0b, Whh0b, Wih1b, Whh1b,
                                      b_ih0, b_hh0, b_ih1, b_hh1,
                                      h0, h1, c0, c1,
                                      k, e0, k - 1, e1);
  }

  // 4) final FC. h1 of step T-1 was written to parity (T)&1 == 0.
  const bf16_t* hf = h1 + (size_t)(0 * 2 + 0) * B * H;
  const bf16_t* hr = h1 + (size_t)(1 * 2 + 0) * B * H;
  fc_kernel<<<8, 256, 0, stream>>>(hf, hr, Wfcb, b_fc, out);
}
